// MambaFlipVlop_60284160967101
// MI455X (gfx1250) — matrix operand order repaired and checked by reading, not compiled, not run on hardware
//
#include <hip/hip_runtime.h>
#include <hip/hip_bf16.h>

#define DM   768
#define DI   1536
#define PP   64
#define HH   24
#define NN   64
#define KK   4
#define CONVD 1664            // DI + 2*N
#define NPROJ 3224            // 2*DI + 2*N + H
#define NPAD  3328            // NPROJ padded to multiple of 128
#define LSEQ  1024
#define BATCH 2
#define ROWS  (BATCH*LSEQ)    // 2048
#define QC   64               // scan chunk length
#define NCH  (LSEQ/QC)        // 16 chunks
#define UNITS (BATCH*HH*NCH)  // 768 per direction

typedef __attribute__((ext_vector_type(16))) _Float16 v16h;
typedef __attribute__((ext_vector_type(8)))  _Float16 v8h;
typedef __attribute__((ext_vector_type(8)))  float    v8f;

__device__ __forceinline__ int seqmap(int k) {
    // SEQ[k] = 1023 - 32*(k%32) - k/32  (involution on [0,1024))
    return 1023 - ((k & 31) << 5) - (k >> 5);
}

__device__ __forceinline__ float block_reduce(float v, float* s) {
    int t = threadIdx.x;
    s[t] = v; __syncthreads();
    for (int off = 128; off > 0; off >>= 1) {
        if (t < off) s[t] += s[t + off];
        __syncthreads();
    }
    float r = s[0];
    __syncthreads();
    return r;
}

// ---------------- WMMA fragment loaders (ISA 7.12.2 16-bit layouts) --------
__device__ __forceinline__ v16h load_a_frag(const _Float16* __restrict__ base, int ld,
                                            int m0, int k0, int lane) {
    int r  = m0 + (lane & 15);
    int hi = (lane >> 4) & 1;
    const _Float16* p = base + (size_t)r * ld + k0 + hi * 8;
    v8h lo = *(const v8h*)p;
    v8h hh = *(const v8h*)(p + 16);
    v16h out;
#pragma unroll
    for (int i = 0; i < 8; ++i) { out[i] = lo[i]; out[i + 8] = hh[i]; }
    return out;
}

__device__ __forceinline__ v16h load_b_frag(const _Float16* __restrict__ base, int ld,
                                            int n0, int k0, int lane) {
    int r  = n0 + (lane & 15);
    int kb = k0 + ((lane >> 4) << 3);
    const _Float16* p = base + (size_t)r * ld + kb;
    v8h lo = *(const v8h*)p;
    v8h hh = *(const v8h*)(p + 16);
    v16h out;
#pragma unroll
    for (int i = 0; i < 8; ++i) { out[i] = lo[i]; out[i + 8] = hh[i]; }
    return out;
}

// ---------------- LayerNorm (+ SEQ gather for backward dir), write f16 -----
__global__ void ln_kernel(const float* __restrict__ x,
                          const float* __restrict__ w0, const float* __restrict__ b0,
                          const float* __restrict__ w1, const float* __restrict__ b1,
                          _Float16* __restrict__ out0, _Float16* __restrict__ out1) {
    __shared__ float sred[256];
    int dir = blockIdx.x >> 11;
    int row = blockIdx.x & 2047;
    int src = dir ? ((row & ~1023) + seqmap(row & 1023)) : row;
    const float* xr = x + (size_t)src * DM;
    int t = threadIdx.x;
    float v0 = xr[t], v1 = xr[t + 256], v2 = xr[t + 512];
    float mean = block_reduce(v0 + v1 + v2, sred) * (1.0f / 768.0f);
    float d0 = v0 - mean, d1 = v1 - mean, d2 = v2 - mean;
    float var = block_reduce(d0*d0 + d1*d1 + d2*d2, sred) * (1.0f / 768.0f);
    float rs = rsqrtf(var + 1e-5f);
    const float* w = dir ? w1 : w0;
    const float* bb = dir ? b1 : b0;
    _Float16* o = (dir ? out1 : out0) + (size_t)row * DM;
    o[t]       = (_Float16)(d0 * rs * w[t]       + bb[t]);
    o[t + 256] = (_Float16)(d1 * rs * w[t + 256] + bb[t + 256]);
    o[t + 512] = (_Float16)(d2 * rs * w[t + 512] + bb[t + 512]);
}

// ---------------- f32 -> f16 cast with zero row padding --------------------
__global__ void cast_pad_kernel(const float* __restrict__ src, _Float16* __restrict__ dst,
                                long rows_src, long total, int cols) {
    long idx = (long)blockIdx.x * 256 + threadIdx.x;
    if (idx >= total) return;
    long r = idx / cols;
    dst[idx] = (r < rows_src) ? (_Float16)src[idx] : (_Float16)0.0f;
}

// ---------------- GEMM: C(MxN,f32) = A(MxK,f16) * B(NxK,f16)^T -------------
__global__ void gemm_f16_kernel(const _Float16* __restrict__ A, const _Float16* __restrict__ B,
                                float* __restrict__ C, int K, int ldc) {
    int wave = threadIdx.x >> 5;
    int lane = threadIdx.x & 31;
    int m0 = (blockIdx.y * 4 + (wave >> 1)) * 16;
    int n0 = (blockIdx.x * 2 + (wave & 1)) * 64;
    v8f acc0 = {}, acc1 = {}, acc2 = {}, acc3 = {};
    for (int k0 = 0; k0 < K; k0 += 32) {
        v16h a  = load_a_frag(A, K, m0, k0, lane);
        v16h b0 = load_b_frag(B, K, n0 +  0, k0, lane);
        acc0 = __builtin_amdgcn_wmma_f32_16x16x32_f16(false, a, false, b0, (short)0, acc0, false, false);
        v16h b1 = load_b_frag(B, K, n0 + 16, k0, lane);
        acc1 = __builtin_amdgcn_wmma_f32_16x16x32_f16(false, a, false, b1, (short)0, acc1, false, false);
        v16h b2 = load_b_frag(B, K, n0 + 32, k0, lane);
        acc2 = __builtin_amdgcn_wmma_f32_16x16x32_f16(false, a, false, b2, (short)0, acc2, false, false);
        v16h b3 = load_b_frag(B, K, n0 + 48, k0, lane);
        acc3 = __builtin_amdgcn_wmma_f32_16x16x32_f16(false, a, false, b3, (short)0, acc3, false, false);
    }
    int nn = lane & 15;
    int mb = m0 + ((lane >> 4) ? 8 : 0);
#pragma unroll
    for (int v = 0; v < 8; ++v) {
        float* cr = C + (size_t)(mb + v) * ldc + n0 + nn;
        cr[0]  = acc0[v];
        cr[16] = acc1[v];
        cr[32] = acc2[v];
        cr[48] = acc3[v];
    }
}

// ---------------- causal depthwise conv (K=4) + bias + SiLU ----------------
__global__ void conv_silu_kernel(const float* __restrict__ zx, const float* __restrict__ cw,
                                 const float* __restrict__ cb, float* __restrict__ xBCa) {
    int idx = blockIdx.x * 256 + threadIdx.x;
    if (idx >= ROWS * CONVD) return;
    int c   = idx % CONVD;
    int row = idx / CONVD;
    int l   = row & 1023;
    const float* src = zx + DI + c;
    float acc = cb[c];
#pragma unroll
    for (int j = 0; j < KK; ++j) {
        int ll = l - 3 + j;
        if (ll >= 0) acc += cw[c * KK + j] * src[(size_t)(row - 3 + j) * NPAD];
    }
    acc = acc / (1.0f + expf(-acc));
    xBCa[idx] = acc;
}

// ---------------- dt = softplus(dt_raw + dt_bias) --------------------------
__global__ void dt_kernel(const float* __restrict__ zx, const float* __restrict__ dt_bias,
                          float* __restrict__ dtb) {
    int idx = blockIdx.x * 256 + threadIdx.x;
    if (idx >= ROWS * HH) return;
    int h = idx % HH, row = idx / HH;
    float v = zx[(size_t)row * NPAD + (2 * DI + 2 * NN) + h] + dt_bias[h];
    dtb[idx] = (v > 20.0f) ? v : log1pf(expf(v));
}

// ================= chunked scan (Mamba2 SSD) ===============================
// unit u = (b*HH + h)*NCH + c ; per-unit 64x64 f16 operand tiles
// ---------------- per-chunk operand prep -----------------------------------
__global__ void chunk_prep_kernel(const float* __restrict__ xBCa, const float* __restrict__ dtb,
                                  const float* __restrict__ Alog,
                                  _Float16* __restrict__ Bc, _Float16* __restrict__ Cc,
                                  _Float16* __restrict__ Cw, _Float16* __restrict__ Bwt,
                                  _Float16* __restrict__ XT,
                                  float* __restrict__ csum_buf, float* __restrict__ beta) {
    __shared__ float sc[64], sL[64], sW[64];
    int u = blockIdx.x;
    int b = u / (HH * NCH), h = (u / NCH) % HH, c = u % NCH;
    int t = threadIdx.x;
    float A = -expf(Alog[h]);
    float dtv = 0.0f;
    if (t < 64) {
        dtv = dtb[(b * LSEQ + c * QC + t) * HH + h];
        sc[t] = dtv * A;
    }
    __syncthreads();
    // inclusive Hillis-Steele scan of a_t over the chunk
    for (int off = 1; off < 64; off <<= 1) {
        float add = 0.0f;
        if (t < 64 && t >= off) add = sc[t - off];
        __syncthreads();
        if (t < 64) sc[t] += add;
        __syncthreads();
    }
    float total = sc[63];
    if (t < 64) {
        float ci = sc[t];
        csum_buf[u * 64 + t] = ci;
        sL[t] = expf(ci);                 // inter-chunk decay for C rows
        sW[t] = expf(total - ci) * dtv;   // chunk-state weights for B rows
    }
    if (t == 0) beta[u] = expf(total);
    __syncthreads();
    size_t base = (size_t)u * 4096;
    for (int idx = t; idx < 4096; idx += 256) {
        int i = idx >> 6, k = idx & 63;
        int rg_i = b * LSEQ + c * QC + i;
        int rg_k = b * LSEQ + c * QC + k;
        float Bv = xBCa[(size_t)rg_i * CONVD + DI + k];
        float Cv = xBCa[(size_t)rg_i * CONVD + DI + NN + k];
        Bc[base + idx]  = (_Float16)Bv;                                        // B [j][n]
        Cc[base + idx]  = (_Float16)Cv;                                        // C [i][n]
        Cw[base + idx]  = (_Float16)(sL[i] * Cv);                              // exp(csum_i)*C
        Bwt[base + idx] = (_Float16)(sW[k] * xBCa[(size_t)rg_k * CONVD + DI + i]); // Bw^T [n][j]
        XT[base + idx]  = (_Float16)(xBCa[(size_t)rg_k * CONVD + h * PP + i]);     // X^T [p][j]
    }
}

// -------- intra-chunk: G = C*B^T (WMMA) -> mask/decay -> Y_intra, T --------
__global__ void intra_kernel(const _Float16* __restrict__ Cc, const _Float16* __restrict__ Bc,
                             const _Float16* __restrict__ XT, const _Float16* __restrict__ Bwt,
                             const float* __restrict__ csum_buf, const float* __restrict__ dtb,
                             float* __restrict__ Yi, float* __restrict__ Tt) {
    __shared__ __align__(16) _Float16 sGm[64 * 64];
    __shared__ float s_cs[64], s_dt[64];
    int u = blockIdx.x;
    int b = u / (HH * NCH), h = (u / NCH) % HH, c = u % NCH;
    int t = threadIdx.x;
    size_t base = (size_t)u * 4096;
    if (t < 64) {
        s_cs[t] = csum_buf[u * 64 + t];
        s_dt[t] = dtb[(b * LSEQ + c * QC + t) * HH + h];
    }
    __syncthreads();
    int wave = t >> 5, lane = t & 31;
    int nn = lane & 15, mb = (lane >> 4) ? 8 : 0;
    // GEMM1: G[i][j] = sum_n C[i][n]*B[j][n]; apply causal decay mask -> LDS
#pragma unroll
    for (int q = 0; q < 2; ++q) {
        int tile = wave * 2 + q;
        int ti = (tile >> 2) * 16, tj = (tile & 3) * 16;
        v8f acc = {};
        for (int k0 = 0; k0 < 64; k0 += 32) {
            v16h a  = load_a_frag(Cc + base, 64, ti, k0, lane);
            v16h bf = load_b_frag(Bc + base, 64, tj, k0, lane);
            acc = __builtin_amdgcn_wmma_f32_16x16x32_f16(false, a, false, bf, (short)0, acc, false, false);
        }
        int j = tj + nn;
        float cj = s_cs[j], dj = s_dt[j];
#pragma unroll
        for (int v = 0; v < 8; ++v) {
            int i = ti + mb + v;
            float g = (j <= i) ? expf(s_cs[i] - cj) * dj * acc[v] : 0.0f;
            sGm[i * 64 + j] = (_Float16)g;
        }
    }
    __syncthreads();
    // GEMM2: Y_intra = Gm @ X  (A from LDS, B = X^T [p][j])
#pragma unroll
    for (int q = 0; q < 2; ++q) {
        int tile = wave * 2 + q;
        int ti = (tile >> 2) * 16, tp = (tile & 3) * 16;
        v8f acc = {};
        for (int k0 = 0; k0 < 64; k0 += 32) {
            v16h a  = load_a_frag((const _Float16*)sGm, 64, ti, k0, lane);
            v16h bf = load_b_frag(XT + base, 64, tp, k0, lane);
            acc = __builtin_amdgcn_wmma_f32_16x16x32_f16(false, a, false, bf, (short)0, acc, false, false);
        }
#pragma unroll
        for (int v = 0; v < 8; ++v)
            Yi[base + (size_t)(ti + mb + v) * 64 + tp + nn] = acc[v];
    }
    // GEMM3: T^T[p][n] = sum_j X^T[p][j] * Bw^T[n][j]
#pragma unroll
    for (int q = 0; q < 2; ++q) {
        int tile = wave * 2 + q;
        int tp = (tile >> 2) * 16, tn = (tile & 3) * 16;
        v8f acc = {};
        for (int k0 = 0; k0 < 64; k0 += 32) {
            v16h a  = load_a_frag(XT + base, 64, tp, k0, lane);
            v16h bf = load_b_frag(Bwt + base, 64, tn, k0, lane);
            acc = __builtin_amdgcn_wmma_f32_16x16x32_f16(false, a, false, bf, (short)0, acc, false, false);
        }
#pragma unroll
        for (int v = 0; v < 8; ++v)
            Tt[base + (size_t)(tp + mb + v) * 64 + tn + nn] = acc[v];
    }
}

// -------- inter-chunk recurrence: only NCH=16 sequential steps -------------
__global__ void chunk_recur_kernel(const float* __restrict__ Tt, const float* __restrict__ beta,
                                   _Float16* __restrict__ Sp) {
    int g = blockIdx.x;            // b*HH + h
    int t = threadIdx.x;
    float S[16];
#pragma unroll
    for (int i = 0; i < 16; ++i) S[i] = 0.0f;
    for (int c = 0; c < NCH; ++c) {
        size_t base = ((size_t)g * NCH + c) * 4096;
        float bt = beta[g * NCH + c];
#pragma unroll
        for (int i = 0; i < 16; ++i) {
            int e = i * 256 + t;
            Sp[base + e] = (_Float16)S[i];        // state BEFORE this chunk, layout [p][n]
            S[i] = bt * S[i] + Tt[base + e];
        }
    }
}

// -------- inter-chunk contribution + combine: Y = Yinter + Yintra + D*x ----
__global__ void inter_kernel(const _Float16* __restrict__ Cw, const _Float16* __restrict__ Sp,
                             const float* __restrict__ Yi, const float* __restrict__ xBCa,
                             const float* __restrict__ Dv, float* __restrict__ ybuf) {
    int u = blockIdx.x;
    int b = u / (HH * NCH), h = (u / NCH) % HH, c = u % NCH;
    int t = threadIdx.x;
    int wave = t >> 5, lane = t & 31;
    int nn = lane & 15, mb = (lane >> 4) ? 8 : 0;
    size_t base = (size_t)u * 4096;
    float Dh = Dv[h];
#pragma unroll
    for (int q = 0; q < 2; ++q) {
        int tile = wave * 2 + q;
        int ti = (tile >> 2) * 16, tp = (tile & 3) * 16;
        v8f acc = {};
        for (int k0 = 0; k0 < 64; k0 += 32) {
            v16h a  = load_a_frag(Cw + base, 64, ti, k0, lane);
            v16h bf = load_b_frag(Sp + base, 64, tp, k0, lane);
            acc = __builtin_amdgcn_wmma_f32_16x16x32_f16(false, a, false, bf, (short)0, acc, false, false);
        }
        int p = tp + nn;
#pragma unroll
        for (int v = 0; v < 8; ++v) {
            int i = ti + mb + v;
            int rg = b * LSEQ + c * QC + i;
            float y = acc[v] + Yi[base + (size_t)i * 64 + p]
                    + Dh * xBCa[(size_t)rg * CONVD + h * PP + p];
            ybuf[(size_t)rg * DI + h * PP + p] = y;
        }
    }
}

// ---------------- gate with silu(z), RMSNorm, cast f16 ---------------------
__global__ void gatenorm_kernel(const float* __restrict__ zx, const float* __restrict__ ybuf,
                                const float* __restrict__ norm_w, _Float16* __restrict__ yh) {
    __shared__ float sred[256];
    int row = blockIdx.x;
    int t = threadIdx.x;
    const float* zr = zx   + (size_t)row * NPAD;
    const float* yr = ybuf + (size_t)row * DI;
    float vals[6];
    float ss = 0.0f;
#pragma unroll
    for (int i = 0; i < 6; ++i) {
        int c = t + i * 256;
        float z = zr[c];
        float y = yr[c] * (z / (1.0f + expf(-z)));
        vals[i] = y;
        ss += y * y;
    }
    float s2 = block_reduce(ss, sred);
    float rs = rsqrtf(s2 * (1.0f / 1536.0f) + 1e-5f);
    _Float16* o = yh + (size_t)row * DI;
#pragma unroll
    for (int i = 0; i < 6; ++i) {
        int c = t + i * 256;
        o[c] = (_Float16)(vals[i] * rs * norm_w[c]);
    }
}

// ---------------- final residual + un-permute ------------------------------
__global__ void final_add_kernel(const float* __restrict__ x, const float* __restrict__ outf,
                                 const float* __restrict__ outb, float* __restrict__ out) {
    int idx = blockIdx.x * 256 + threadIdx.x;
    if (idx >= ROWS * DM) return;
    int d = idx % DM;
    int row = idx / DM;
    int l = row & 1023, bb = row >> 10;
    int s1 = seqmap(l), s2 = seqmap(s1);
    out[idx] = x[idx] + outf[idx]
             + x[(size_t)((bb << 10) + s2) * DM + d]
             + outb[(size_t)((bb << 10) + s1) * DM + d];
}

extern "C" void kernel_launch(void* const* d_in, const int* in_sizes, int n_in,
                              void* d_out, int out_size, void* d_ws, size_t ws_size,
                              hipStream_t stream) {
    const float* x = (const float*)d_in[0];
    const float* ln_w[2]   = { (const float*)d_in[1],  (const float*)d_in[11] };
    const float* ln_b[2]   = { (const float*)d_in[2],  (const float*)d_in[12] };
    const float* in_w[2]   = { (const float*)d_in[3],  (const float*)d_in[13] };
    const float* conv_w[2] = { (const float*)d_in[4],  (const float*)d_in[14] };
    const float* conv_b[2] = { (const float*)d_in[5],  (const float*)d_in[15] };
    const float* dt_bias[2]= { (const float*)d_in[6],  (const float*)d_in[16] };
    const float* A_log[2]  = { (const float*)d_in[7],  (const float*)d_in[17] };
    const float* Dp[2]     = { (const float*)d_in[8],  (const float*)d_in[18] };
    const float* norm_w[2] = { (const float*)d_in[9],  (const float*)d_in[19] };
    const float* out_w[2]  = { (const float*)d_in[10], (const float*)d_in[20] };

    char* p = (char*)d_ws;
    auto carve = [&](size_t bytes) { void* r = (void*)p; p += (bytes + 255) & ~(size_t)255; return r; };
    _Float16 *xn16[2], *win16[2], *yh[2], *wout16[2];
    _Float16 *Bc[2], *Cc[2], *Cw[2], *Bwt[2], *XT[2], *Sp[2];
    float *zx[2], *xBCa[2], *dtb[2], *ybuf[2], *outd[2], *csum[2], *beta[2], *Yi[2], *Tt[2];
    for (int d = 0; d < 2; ++d) {
        xn16[d]   = (_Float16*)carve((size_t)ROWS * DM * 2);
        win16[d]  = (_Float16*)carve((size_t)NPAD * DM * 2);
        zx[d]     = (float*)   carve((size_t)ROWS * NPAD * 4);
        xBCa[d]   = (float*)   carve((size_t)ROWS * CONVD * 4);
        dtb[d]    = (float*)   carve((size_t)ROWS * HH * 4);
        ybuf[d]   = (float*)   carve((size_t)ROWS * DI * 4);
        yh[d]     = (_Float16*)carve((size_t)ROWS * DI * 2);
        wout16[d] = (_Float16*)carve((size_t)DM * DI * 2);
        outd[d]   = (float*)   carve((size_t)ROWS * DM * 4);
        Bc[d]     = (_Float16*)carve((size_t)UNITS * 4096 * 2);
        Cc[d]     = (_Float16*)carve((size_t)UNITS * 4096 * 2);
        Cw[d]     = (_Float16*)carve((size_t)UNITS * 4096 * 2);
        Bwt[d]    = (_Float16*)carve((size_t)UNITS * 4096 * 2);
        XT[d]     = (_Float16*)carve((size_t)UNITS * 4096 * 2);
        Sp[d]     = (_Float16*)carve((size_t)UNITS * 4096 * 2);
        Yi[d]     = (float*)   carve((size_t)UNITS * 4096 * 4);
        Tt[d]     = (float*)   carve((size_t)UNITS * 4096 * 4);
        csum[d]   = (float*)   carve((size_t)UNITS * 64 * 4);
        beta[d]   = (float*)   carve((size_t)UNITS * 4);
    }

    // 1) LayerNorm (fwd) + LayerNorm of SEQ-gathered x (bwd), f16 output
    ln_kernel<<<4096, 256, 0, stream>>>(x, ln_w[0], ln_b[0], ln_w[1], ln_b[1], xn16[0], xn16[1]);

    // 2) weight casts (in_w padded 3224 -> 3328 rows with zeros)
    for (int d = 0; d < 2; ++d) {
        long tin = (long)NPAD * DM;
        cast_pad_kernel<<<(unsigned)((tin + 255) / 256), 256, 0, stream>>>(in_w[d],  win16[d],  NPROJ, tin, DM);
        long tout = (long)DM * DI;
        cast_pad_kernel<<<(unsigned)((tout + 255) / 256), 256, 0, stream>>>(out_w[d], wout16[d], DM,   tout, DI);
    }

    // 3) in-projection GEMM (M=2048, N=3328pad, K=768)
    for (int d = 0; d < 2; ++d)
        gemm_f16_kernel<<<dim3(NPAD / 128, ROWS / 64), 256, 0, stream>>>(xn16[d], win16[d], zx[d], DM, NPAD);

    // 4) conv+SiLU, dt softplus, then chunked SSD scan
    for (int d = 0; d < 2; ++d) {
        conv_silu_kernel<<<(ROWS * CONVD + 255) / 256, 256, 0, stream>>>(zx[d], conv_w[d], conv_b[d], xBCa[d]);
        dt_kernel<<<(ROWS * HH + 255) / 256, 256, 0, stream>>>(zx[d], dt_bias[d], dtb[d]);
        chunk_prep_kernel<<<UNITS, 256, 0, stream>>>(xBCa[d], dtb[d], A_log[d],
                                                     Bc[d], Cc[d], Cw[d], Bwt[d], XT[d],
                                                     csum[d], beta[d]);
        intra_kernel<<<UNITS, 256, 0, stream>>>(Cc[d], Bc[d], XT[d], Bwt[d],
                                                csum[d], dtb[d], Yi[d], Tt[d]);
        chunk_recur_kernel<<<BATCH * HH, 256, 0, stream>>>(Tt[d], beta[d], Sp[d]);
        inter_kernel<<<UNITS, 256, 0, stream>>>(Cw[d], Sp[d], Yi[d], xBCa[d], Dp[d], ybuf[d]);
    }

    // 5) gate + RMSNorm -> f16 ; out-projection GEMM (M=2048, N=768, K=1536)
    for (int d = 0; d < 2; ++d) {
        gatenorm_kernel<<<ROWS, 256, 0, stream>>>(zx[d], ybuf[d], norm_w[d], yh[d]);
        gemm_f16_kernel<<<dim3(DM / 128, ROWS / 64), 256, 0, stream>>>(yh[d], wout16[d], outd[d], DI, DM);
    }

    // 6) residual + un-permute backward direction
    final_add_kernel<<<(ROWS * DM + 255) / 256, 256, 0, stream>>>(x, outd[0], outd[1], (float*)d_out);
}
